// Multi_GINs_68247030333449
// MI455X (gfx1250) — matrix operand order repaired and checked by reading, not compiled, not run on hardware
//
#include <hip/hip_runtime.h>
#include <hip/hip_bf16.h>

// ---------------------------------------------------------------------------
// Multi-step GNN+cross-attention pipeline for MI455X (gfx1250, wave32).
// Dense GEMMs on v_wmma_f32_16x16x32_bf16 (f32 -> bf16 convert on the fly,
// f32 accumulate). All global loads unconditional (clamped rows) so they
// lower to global_load_b128; B fragments are two 16B LDS reads (k +0..7, +16..23).
// ---------------------------------------------------------------------------

#define D   128
#define NH  4
#define DKH 32

typedef __attribute__((ext_vector_type(16))) __bf16 v16bf;
typedef __attribute__((ext_vector_type(8)))  float  v8f;

__device__ __forceinline__ v8f wmma_bf16(v16bf a, v16bf b, v8f c) {
  return __builtin_amdgcn_wmma_f32_16x16x32_bf16(
      false, a, false, b, (short)0, c, false, false);
}

__device__ __forceinline__ void pack8(v16bf& a, int base, float4 f0, float4 f1) {
  a[base + 0] = (__bf16)f0.x; a[base + 1] = (__bf16)f0.y;
  a[base + 2] = (__bf16)f0.z; a[base + 3] = (__bf16)f0.w;
  a[base + 4] = (__bf16)f1.x; a[base + 5] = (__bf16)f1.y;
  a[base + 6] = (__bf16)f1.z; a[base + 7] = (__bf16)f1.w;
}

// CDNA5 16-bit A-matrix 16x32 lane layout: lane L holds row M = L%16;
// half = L/16: elements 0..7 -> K = 8*half+0..7, 8..15 -> K = 16+8*half+0..7.
// `p` must point at rowBase + kt*32 + 8*half (32B aligned) -> 4x b128 loads.
__device__ __forceinline__ v16bf load_a_f32(const float* __restrict__ p) {
  const float4* q = (const float4*)p;
  v16bf a;
  pack8(a, 0, q[0], q[1]);      // K offsets +0..7
  pack8(a, 8, q[4], q[5]);      // K offsets +16..23
  return a;
}

// bf16 p[0..7] and p[16..23] (16B-aligned) from LDS -> two ds_load_b128.
__device__ __forceinline__ v16bf load_b_lds(const __bf16* __restrict__ p) {
  union { uint4 u[2]; v16bf v; } c;
  c.u[0] = ((const uint4*)p)[0];
  c.u[1] = ((const uint4*)p)[2];
  return c.v;
}

// ---------------------------------------------------------------------------
// trans(v): out[i,:] = v[i] * trans_w + trans_b
// ---------------------------------------------------------------------------
__global__ void k_trans(const float* __restrict__ v,
                        const float* __restrict__ tw,
                        const float* __restrict__ tb,
                        float* __restrict__ out, int n) {
  int i = blockIdx.x;
  int j = threadIdx.x;
  if (i < n) out[(long)i * D + j] = v[i] * tw[j] + tb[j];
}

__global__ void k_zero(float* __restrict__ p, long n) {
  long i = (long)blockIdx.x * blockDim.x + threadIdx.x;
  if (i < n) p[i] = 0.f;
}

// ---------------------------------------------------------------------------
// COO SpMM scatter: out[row] += val * y[col]   (one wave per edge)
// ---------------------------------------------------------------------------
__global__ void k_spmm(const int* __restrict__ row, const int* __restrict__ col,
                       const float* __restrict__ val,
                       const float* __restrict__ y, float* __restrict__ out,
                       int e) {
  int wave = blockIdx.x * (blockDim.x >> 5) + (threadIdx.x >> 5);
  int lane = threadIdx.x & 31;
  if (wave >= e) return;
  int r = row[wave], c = col[wave];
  float vv = val[wave];
  float4 d = ((const float4*)(y + (long)c * D))[lane];
  float* dst = out + (long)r * D + lane * 4;
  atomicAdd(dst + 0, vv * d.x);
  atomicAdd(dst + 1, vv * d.y);
  atomicAdd(dst + 2, vv * d.z);
  atomicAdd(dst + 3, vv * d.w);
}

// ---------------------------------------------------------------------------
// GEMM: Y[M,N] = act(X[M,K] @ W[K,N] + bias).
// W staged TRANSPOSED in LDS as bf16 [N][K] so B fragments are contiguous.
// 256 threads = 8 waves; block = 128 rows x all N. K is compile-time so the
// per-wave A fragments (K/32 x v16bf) stay fully in registers.
// ---------------------------------------------------------------------------
template <int K, bool FULL>
__global__ __launch_bounds__(256)
void k_gemm(const float* __restrict__ X, const float* __restrict__ W,
            const float* __restrict__ bias, float* __restrict__ Y,
            int M, int N, int act) {
  extern __shared__ char smem[];
  __bf16* wlds = (__bf16*)smem;                       // [N][K]
  float*  blds = (float*)(smem + (size_t)K * N * 2);  // [N]

  int tid = threadIdx.x;
  int nsh = (N == 256) ? 8 : 7;                       // N is a power of two
  int nmask = N - 1;
  for (int idx = tid; idx < K * N; idx += 256) {
    int k = idx >> nsh, nn = idx & nmask;             // coalesced global read
    wlds[nn * K + k] = (__bf16)W[idx];
  }
  for (int idx = tid; idx < N; idx += 256)
    blds[idx] = bias ? bias[idx] : 0.f;
  __syncthreads();

  int wid  = tid >> 5;
  int lane = tid & 31;
  int half = lane >> 4;
  int lm   = lane & 15;
  int blockRow = blockIdx.x * 128;
  int arow = blockRow + wid * 16 + lm;                // A row = lane%16
  if (!FULL) arow = min(arow, M - 1);                 // clamp: finite data
  const float* xrow = X + (long)arow * K + 8 * half;

  constexpr int KT = K / 32;
  v16bf afrag[KT];
#pragma unroll
  for (int kt = 0; kt < KT; ++kt)
    afrag[kt] = load_a_f32(xrow + kt * 32);           // 4x global_load_b128

  int ntiles = N >> 4;
  for (int nt = 0; nt < ntiles; ++nt) {
    v8f acc;
    float bv = blds[nt * 16 + lm];                    // C col = lane%16
#pragma unroll
    for (int v = 0; v < 8; ++v) acc[v] = bv;

    const __bf16* wp = wlds + (long)(nt * 16 + lm) * K + 8 * half;
#pragma unroll
    for (int kt = 0; kt < KT; ++kt) {
      v16bf b = load_b_lds(wp + kt * 32);             // 2x ds_load_b128
      acc = wmma_bf16(afrag[kt], b, acc);
    }
    // C/D layout: VGPR v -> row M = v + 8*half, col = lane%16
#pragma unroll
    for (int v = 0; v < 8; ++v) {
      int orow = blockRow + wid * 16 + v + 8 * half;
      float x = acc[v];
      if (act == 1) x = (x > 0.f) ? x : 0.01f * x;
      if (FULL || orow < M)
        Y[(long)orow * N + nt * 16 + lm] = x;
    }
  }
}

// ---------------------------------------------------------------------------
// Cross-attention, softmax over the QUERY axis (per reference):
//   s[n,q] = (k[n,:].q[q,:])/sqrt(dk);  w = softmax_q(s);  out = w @ v + x
// 2 waves per block, one head each. Scores in LDS [16 x mPad] per wave.
// ---------------------------------------------------------------------------
__global__ __launch_bounds__(64)
void k_attn(const float* __restrict__ Kb, const float* __restrict__ Qb,
            const float* __restrict__ Vb, float* __restrict__ X,
            int n, int m, int mPad) {
  extern __shared__ float smemf[];
  int tid  = threadIdx.x;
  int wid  = tid >> 5;
  int lane = tid & 31;
  int half = lane >> 4;
  int lm   = lane & 15;
  int tile = blockIdx.x;
  int head = blockIdx.y * 2 + wid;
  float* sw = smemf + wid * 16 * mPad;
  const float scale = 0.17677669529663687f;           // 1/sqrt(32)

  int row  = min(tile * 16 + lm, n - 1);              // clamped: finite data
  // K-rows fragment for this head (dk=32 -> one WMMA K-step)
  v16bf ak = load_a_f32(Kb + (long)row * D + head * DKH + 8 * half);

  int qtiles = mPad >> 4;
  for (int qt = 0; qt < qtiles; ++qt) {
    int q = min(qt * 16 + lm, m - 1);                 // clamped query row
    const float4* qp = (const float4*)(Qb + (long)q * D + head * DKH + 8 * half);
    v16bf b;                                          // B[k][q]=Q[q,head*32+k]
    pack8(b, 0, qp[0], qp[1]);
    pack8(b, 8, qp[4], qp[5]);
    v8f s = {};
    s = wmma_bf16(ak, b, s);
#pragma unroll
    for (int v = 0; v < 8; ++v) {
      int c = qt * 16 + lm;
      sw[(v + 8 * half) * mPad + c] = (c < m) ? s[v] * scale : -3.0e38f;
    }
  }
  __syncthreads();

  // softmax over queries: lanes 0..15 own one local row each
  if (lane < 16) {
    float* rp = sw + lane * mPad;
    float mx = -3.0e38f;
    for (int c = 0; c < m; ++c) mx = fmaxf(mx, rp[c]);
    float sum = 0.f;
    for (int c = 0; c < m; ++c) { float e = __expf(rp[c] - mx); rp[c] = e; sum += e; }
    float inv = 1.f / sum;
    for (int c = 0; c < m; ++c) rp[c] *= inv;
    for (int c = m; c < mPad; ++c) rp[c] = 0.f;       // padded K-cols -> 0
  }
  __syncthreads();

  // out[16,32] = W[16,mPad] @ V[mPad,32]; two 16-col tiles; residual add
  int ktiles = mPad >> 5;
  bool fullTile = (tile * 16 + 16) <= n;
#pragma unroll
  for (int nt = 0; nt < 2; ++nt) {
    v8f acc = {};
    for (int kt = 0; kt < ktiles; ++kt) {
      const float4* wp = (const float4*)(sw + lm * mPad + kt * 32 + 8 * half);
      v16bf aw;                                       // A from LDS score rows
      pack8(aw, 0, wp[0], wp[1]);
      pack8(aw, 8, wp[4], wp[5]);
      v16bf bv;                                       // B[k][c]=V[k,head*32+...]
#pragma unroll
      for (int e = 0; e < 16; ++e) {
        int k = min(kt * 32 + 8 * half + e + (e & 8), m - 1);  // clamped; w==0 past m
        bv[e] = (__bf16)Vb[(long)k * D + head * DKH + nt * 16 + lm];
      }
      acc = wmma_bf16(aw, bv, acc);
    }
#pragma unroll
    for (int v = 0; v < 8; ++v) {
      int orow = tile * 16 + v + 8 * half;
      long off = (long)orow * D + head * DKH + nt * 16 + lm;
      if (fullTile) X[off] += acc[v];                 // uniform fast path
      else if (orow < n) X[off] += acc[v];
    }
  }
}

// ---------------------------------------------------------------------------
// Host-side orchestration
// ---------------------------------------------------------------------------
extern "C" void kernel_launch(void* const* d_in, const int* in_sizes, int n_in,
                              void* d_out, int out_size, void* d_ws, size_t ws_size,
                              hipStream_t stream) {
  (void)in_sizes; (void)n_in; (void)out_size; (void)ws_size;
  const float* trans_w = (const float*)d_in[0];
  const float* trans_b = (const float*)d_in[1];
  const float* g_w  = (const float*)d_in[2];
  const float* g_b  = (const float*)d_in[3];
  const float* qg_w = (const float*)d_in[4];
  const float* qg_b = (const float*)d_in[5];
  const float* q_w  = (const float*)d_in[6];
  const float* k_w  = (const float*)d_in[7];
  const float* v_w  = (const float*)d_in[8];
  const float* a1_w = (const float*)d_in[9];
  const float* a1_b = (const float*)d_in[10];
  const float* a2_w = (const float*)d_in[11];
  const float* a2_b = (const float*)d_in[12];
  const float* x0   = (const float*)d_in[13];
  const float* q0   = (const float*)d_in[14];
  const float* xin[3] = {(const float*)d_in[15], (const float*)d_in[17], (const float*)d_in[19]};
  const float* qin[3] = {(const float*)d_in[16], (const float*)d_in[18], (const float*)d_in[20]};

  const int NMAX = 100000, MMAX = 400;
  const int ns[3] = {50000, 75000, 100000};
  const int ms[3] = {200, 300, 400};

  // Raise dynamic-LDS caps (64KB W stage + bias; 52KB attention scores).
  static bool attrDone = false;
  if (!attrDone) {
    attrDone = true;
    hipFuncSetAttribute((const void*)k_gemm<128, true>,
                        hipFuncAttributeMaxDynamicSharedMemorySize, 132 * 1024);
    hipFuncSetAttribute((const void*)k_gemm<128, false>,
                        hipFuncAttributeMaxDynamicSharedMemorySize, 132 * 1024);
    hipFuncSetAttribute((const void*)k_gemm<256, true>,
                        hipFuncAttributeMaxDynamicSharedMemorySize, 132 * 1024);
    hipFuncSetAttribute((const void*)k_gemm<256, false>,
                        hipFuncAttributeMaxDynamicSharedMemorySize, 132 * 1024);
    hipFuncSetAttribute((const void*)k_attn,
                        hipFuncAttributeMaxDynamicSharedMemorySize, 96 * 1024);
  }

  float* Xb  = (float*)d_out;                 // x region doubles as work buf
  float* QXb = Xb + (long)NMAX * D;

  char* ws = (char*)d_ws;
  auto take = [&](size_t bytes) {
    char* p = ws; ws += (bytes + 255) & ~(size_t)255; return (float*)p;
  };
  float* Y   = take((size_t)NMAX * D * 4);
  float* X2  = take((size_t)NMAX * D * 4);
  float* T   = take((size_t)NMAX * 2 * D * 4);
  float* Yq  = take((size_t)MMAX * D * 4);
  float* X2q = take((size_t)MMAX * D * 4);
  float* Tq  = take((size_t)MMAX * 2 * D * 4);
  float* Qb  = take((size_t)MMAX * D * 4);
  float* Vb  = take((size_t)MMAX * D * 4);

  auto gemm = [&](const float* Xp, const float* Wp, const float* bp, float* Yp,
                  int M, int N, int K, int act) {
    dim3 g((M + 127) / 128);
    size_t sh = (size_t)K * N * 2 + (size_t)N * 4;
    bool full = (M & 127) == 0;
    if (K == 128) {
      if (full) k_gemm<128, true ><<<g, 256, sh, stream>>>(Xp, Wp, bp, Yp, M, N, act);
      else      k_gemm<128, false><<<g, 256, sh, stream>>>(Xp, Wp, bp, Yp, M, N, act);
    } else {
      if (full) k_gemm<256, true ><<<g, 256, sh, stream>>>(Xp, Wp, bp, Yp, M, N, act);
      else      k_gemm<256, false><<<g, 256, sh, stream>>>(Xp, Wp, bp, Yp, M, N, act);
    }
  };
  auto zero = [&](float* p, long len) {
    k_zero<<<(unsigned)((len + 255) / 256), 256, 0, stream>>>(p, len);
  };

  k_trans<<<25000, D, 0, stream>>>(x0, trans_w, trans_b, Xb, 25000);
  k_trans<<<100,   D, 0, stream>>>(q0, trans_w, trans_b, QXb, 100);

  int nprev = 25000, mprev = 100;
  for (int s = 0; s < 3; ++s) {
    int n = ns[s], m = ms[s];
    const float* aval = (const float*)d_in[21 + 6 * s];
    const int*   arow = (const int*)  d_in[22 + 6 * s];
    const int*   acol = (const int*)  d_in[23 + 6 * s];
    const float* qval = (const float*)d_in[24 + 6 * s];
    const int*   qrow = (const int*)  d_in[25 + 6 * s];
    const int*   qcol = (const int*)  d_in[26 + 6 * s];

    k_trans<<<n - nprev, D, 0, stream>>>(xin[s], trans_w, trans_b,
                                         Xb + (long)nprev * D, n - nprev);
    k_trans<<<m - mprev, D, 0, stream>>>(qin[s], trans_w, trans_b,
                                         QXb + (long)mprev * D, m - mprev);

    // ---- x branch: gemm -> spmm -> arrg ----
    gemm(Xb, g_w, g_b, Y, n, D, D, 0);
    zero(X2, (long)n * D);
    int e = 16 * n;
    k_spmm<<<(e + 7) / 8, 256, 0, stream>>>(arow, acol, aval, Y, X2, e);
    gemm(X2, a1_w, a1_b, T, n, 2 * D, D, 1);
    gemm(T, a2_w, a2_b, Xb, n, D, 2 * D, 0);

    // ---- qx branch ----
    gemm(QXb, qg_w, qg_b, Yq, m, D, D, 0);
    zero(X2q, (long)m * D);
    int qe = 16 * m;
    k_spmm<<<(qe + 7) / 8, 256, 0, stream>>>(qrow, qcol, qval, Yq, X2q, qe);
    gemm(X2q, a1_w, a1_b, Tq, m, 2 * D, D, 1);
    gemm(Tq, a2_w, a2_b, QXb, m, D, 2 * D, 0);

    // ---- cross attention (updates x in place with residual) ----
    gemm(Xb,  k_w, nullptr, Y,  n, D, D, 0);
    gemm(QXb, q_w, nullptr, Qb, m, D, D, 0);
    gemm(QXb, v_w, nullptr, Vb, m, D, D, 0);
    int mPad = (m + 31) & ~31;
    dim3 ag((n + 15) / 16, NH / 2);
    k_attn<<<ag, 64, (size_t)2 * 16 * mPad * 4, stream>>>(Y, Qb, Vb, Xb, n, m, mPad);

    nprev = n; mprev = m;
  }
}
